// MultiHeadAttention_82489141887161
// MI455X (gfx1250) — matrix operand order repaired and checked by reading, not compiled, not run on hardware
//
#include <hip/hip_runtime.h>

#define H     16
#define DK    64
#define BS    2
#define NSEQ  2048
#define LDIM  1024
#define EPS   1e-5f

typedef __attribute__((ext_vector_type(16))) __bf16        v16bf;
typedef __attribute__((ext_vector_type(8)))  float         v8f;
typedef __attribute__((ext_vector_type(4)))  float         f4v;
typedef __attribute__((ext_vector_type(4)))  unsigned int  u4v;
typedef __attribute__((ext_vector_type(2)))  float         f2v;
typedef __attribute__((ext_vector_type(2)))  __bf16        bf2v;
typedef __attribute__((ext_vector_type(4)))  unsigned int  v4u;
typedef __attribute__((ext_vector_type(8)))  int           v8i;
typedef __attribute__((ext_vector_type(4)))  int           v4i;

#if __has_builtin(__builtin_amdgcn_tensor_load_to_lds) && __has_builtin(__builtin_amdgcn_s_wait_tensorcnt)
#define HAVE_TDM 1
#else
#define HAVE_TDM 0
#endif

union BF16Frag { v16bf v; unsigned int u[8]; };

__device__ __forceinline__ unsigned int pack_bf16_2(float lo, float hi) {
  f2v f; f[0] = lo; f[1] = hi;
  bf2v b = __builtin_convertvector(f, bf2v);      // v_cvt_pk_bf16_f32
  return __builtin_bit_cast(unsigned int, b);
}
__device__ __forceinline__ u4v pack_bf16_8(const float* t) {
  u4v p;
  p[0] = pack_bf16_2(t[0], t[1]);
  p[1] = pack_bf16_2(t[2], t[3]);
  p[2] = pack_bf16_2(t[4], t[5]);
  p[3] = pack_bf16_2(t[6], t[7]);
  return p;
}
__device__ __forceinline__ unsigned short f32_to_bf16(float f) {
  unsigned int u = __builtin_bit_cast(unsigned int, f);
  u += 0x7FFFu + ((u >> 16) & 1u);
  return (unsigned short)(u >> 16);
}

#if HAVE_TDM
// 2D bf16 tile DMA: global -> LDS with row padding (ISA 8.3/8.4 D# layout).
// pad_icode: 3 -> pad after 16 DWORDs, 4 -> after 32 DWORDs. pad_acode: 3 -> 4 DWORDs.
__device__ __forceinline__ void tdm_load_2d(const void* gptr, unsigned lds_off,
                                            unsigned tile_d0, unsigned tile_d1,
                                            unsigned tensor_d0, unsigned tensor_d1,
                                            unsigned stride_d0,
                                            unsigned pad_icode, unsigned pad_acode) {
  const unsigned long long ga = (unsigned long long)gptr;
  v4u g0;
  g0[0] = 1u;                                            // count=1, user descriptor
  g0[1] = lds_off;                                       // LDS byte address
  g0[2] = (unsigned)(ga & 0xFFFFFFFFu);                  // global_addr[31:0]
  g0[3] = (unsigned)((ga >> 32) & 0x01FFFFFFu) | (2u << 30);  // addr[56:32] | type=2
  v8i g1;
  g1[0] = (int)((1u << 16) | (1u << 20) | (pad_icode << 22) | (pad_acode << 25)); // 2B, pad_en
  g1[1] = (int)((tensor_d0 & 0xFFFFu) << 16);
  g1[2] = (int)(((tensor_d0 >> 16) & 0xFFFFu) | ((tensor_d1 & 0xFFFFu) << 16));
  g1[3] = (int)(((tensor_d1 >> 16) & 0xFFFFu) | (tile_d0 << 16));
  g1[4] = (int)tile_d1;                                  // tile_dim1, tile_dim2=0
  g1[5] = (int)stride_d0;                                // tensor_dim0_stride lo32
  g1[6] = 0;
  g1[7] = 0;
  const v4i z4 = {};
  const v8i z8 = {};
  __builtin_amdgcn_tensor_load_to_lds(g0, g1, z4, z4, z8, 0);
}
#endif

// A-matrix 16x32 bf16 fragment per ISA 7.12.2 (lane&15 = M row;
// VGPR j holds K pair: j<4 -> 2j+8*hi ; j>=4 -> 16+2(j-4)+8*hi).
__device__ __forceinline__ v16bf load_a_frag(const unsigned short* base, int stride, int lane) {
  BF16Frag f;
  const int row = lane & 15;
  const int hi  = (lane >> 4) & 1;
  const unsigned short* rp = base + row * stride;
#pragma unroll
  for (int j = 0; j < 8; ++j) {
    const int kb = ((j < 4) ? (2 * j) : (2 * j + 8)) + 8 * hi;
    f.u[j] = *(const unsigned int*)(rp + kb);
  }
  return f.v;
}

// B-matrix 32x16 bf16 fragment (lane&15 = N col; VGPR j holds K pair:
// j<4 -> 2j+8*hi ; j>=4 -> 16+2(j-4)+8*hi). LDS layout base[col*stride + k].
__device__ __forceinline__ v16bf load_b_frag(const unsigned short* base, int stride, int lane) {
  BF16Frag f;
  const int col = lane & 15;
  const int hi  = (lane >> 4) & 1;
  const unsigned short* cp = base + col * stride + 8 * hi;
#pragma unroll
  for (int j = 0; j < 8; ++j) f.u[j] = *(const unsigned int*)(cp + ((j < 4) ? (2 * j) : (2 * j + 8)));
  return f.v;
}

__device__ __forceinline__ v8f wmma_bf16(v16bf a, v16bf b, v8f c) {
  return __builtin_amdgcn_wmma_f32_16x16x32_bf16(false, a, false, b, (short)0, c, false, false);
}

// ---------------------------------------------------------------------------
// Kernel 1: QKV projection X(4096x1024)@W(1024x1024)+b -> bf16 outputs.
// Q,K stored (b,h,n,d); V stored transposed (b,h,d,n) with packed b128 stores.
// Block 256 thr / 8 waves; 128x64 block tile; 2x2 16x16 tiles per wave.
// ---------------------------------------------------------------------------
#define SA 40   // LDS k-stride (32 + 8 pad)

__global__ __launch_bounds__(256) void proj_kernel(
    const float* __restrict__ q, const float* __restrict__ k, const float* __restrict__ v,
    const float* __restrict__ Wq, const float* __restrict__ bq,
    const float* __restrict__ Wk, const float* __restrict__ bk,
    const float* __restrict__ Wv, const float* __restrict__ bv,
    unsigned short* __restrict__ Qp, unsigned short* __restrict__ Kp,
    unsigned short* __restrict__ Vt) {
  const float *X, *W, *bias;
  if (blockIdx.z == 0)      { X = q; W = Wq; bias = bq; }
  else if (blockIdx.z == 1) { X = k; W = Wk; bias = bk; }
  else                      { X = v; W = Wv; bias = bv; }

  __shared__ __align__(16) unsigned short As[128 * SA];  // [row][k]
  __shared__ __align__(16) unsigned short Bs[64 * SA];   // [col][k]

  const int tid  = threadIdx.x;
  const int lane = tid & 31, wave = tid >> 5;
  const int tr0  = (wave & 3) * 2;
  const int tc0  = (wave >> 2) * 2;
  const int mbase = blockIdx.y * 128;
  const int nbase = blockIdx.x * 64;

  v8f acc00 = {}, acc01 = {}, acc10 = {}, acc11 = {};

  for (int k0 = 0; k0 < LDIM; k0 += 32) {
    __syncthreads();
    { // A tile 128x32 fp32 -> bf16
      const int row = tid >> 1, kg = (tid & 1) * 16;
      const float* src = X + (size_t)(mbase + row) * LDIM + k0 + kg;
      float t[16];
#pragma unroll
      for (int i = 0; i < 4; ++i) *(f4v*)&t[i * 4] = *(const f4v*)(src + i * 4);
      *(u4v*)&As[row * SA + kg]     = pack_bf16_8(&t[0]);
      *(u4v*)&As[row * SA + kg + 8] = pack_bf16_8(&t[8]);
    }
    { // B tile 32x64 register transpose
      const int col = tid & 63, kg = (tid >> 6) * 8;
      const float* src = W + (size_t)(k0 + kg) * LDIM + nbase + col;
      float t[8];
#pragma unroll
      for (int i = 0; i < 8; ++i) t[i] = src[(size_t)i * LDIM];
      *(u4v*)&Bs[col * SA + kg] = pack_bf16_8(t);
    }
    if (k0 + 32 < LDIM) {
      if (tid < 128)
        __builtin_prefetch(X + (size_t)(mbase + tid) * LDIM + k0 + 32, 0, 1);
      else {
        const int t2 = tid - 128;
        __builtin_prefetch(W + (size_t)(k0 + 32 + (t2 >> 1)) * LDIM + nbase + (t2 & 1) * 32, 0, 1);
      }
    }
    __syncthreads();
    v16bf a0 = load_a_frag(&As[(tr0 * 16) * SA], SA, lane);
    v16bf a1 = load_a_frag(&As[((tr0 + 1) * 16) * SA], SA, lane);
    v16bf b0 = load_b_frag(&Bs[(tc0 * 16) * SA], SA, lane);
    v16bf b1 = load_b_frag(&Bs[((tc0 + 1) * 16) * SA], SA, lane);
    acc00 = wmma_bf16(a0, b0, acc00);
    acc01 = wmma_bf16(a0, b1, acc01);
    acc10 = wmma_bf16(a1, b0, acc10);
    acc11 = wmma_bf16(a1, b1, acc11);
  }

  const int col0 = lane & 15, hi = (lane >> 4) & 1;
  const int bidx = mbase >> 11;             // whole 128-row tile in one batch
  const v8f accs[4] = {acc00, acc01, acc10, acc11};
#pragma unroll
  for (int rt = 0; rt < 2; ++rt)
#pragma unroll
    for (int ct = 0; ct < 2; ++ct) {
      const v8f acc = accs[rt * 2 + ct];
      const int c  = nbase + (tc0 + ct) * 16 + col0;
      const int hh = c >> 6, d = c & 63;
      const float bb = bias[c];
      if (blockIdx.z == 2) {
        // V: lane's 8 rows are consecutive n -> one b128 store, transposed layout
        const int n0 = (mbase & (NSEQ - 1)) + (tr0 + rt) * 16 + 8 * hi;
        float t[8];
#pragma unroll
        for (int r = 0; r < 8; ++r) t[r] = acc[r] + bb;
        *(u4v*)&Vt[((size_t)(bidx * H + hh) * DK + d) * NSEQ + n0] = pack_bf16_8(t);
      } else {
        unsigned short* dst = (blockIdx.z == 0) ? Qp : Kp;
#pragma unroll
        for (int r = 0; r < 8; ++r) {
          const int grow = mbase + (tr0 + rt) * 16 + r + 8 * hi;
          const int n    = grow & (NSEQ - 1);
          dst[(((size_t)(bidx * H + hh) * NSEQ + n) << 6) + d] = f32_to_bf16(acc[r] + bb);
        }
      }
    }
}

// ---------------------------------------------------------------------------
// Kernel 2: attention per (b, h, 64-query tile). Two-pass online softmax.
// bf16 Q/K/V tiles DMA'd to LDS by the Tensor Data Mover (wave 0 issues,
// s_wait_tensorcnt + barrier publishes); cooperative-copy fallback otherwise.
// ---------------------------------------------------------------------------
#define SQ 72   // bf16 tile k-stride (64 + 8 pad = 36 DWORDs)
#define SS 66   // f32 score-tile stride

__global__ __launch_bounds__(256) void attn_kernel(
    const unsigned short* __restrict__ Qp, const unsigned short* __restrict__ Kp,
    const unsigned short* __restrict__ Vt,
    float* __restrict__ att, unsigned short* __restrict__ Abf) {
  __shared__ __align__(16) unsigned short Qs[64 * SQ];  // [qrow][d]
  __shared__ __align__(16) unsigned short Ks[64 * SQ];  // [key][d]
  __shared__ __align__(16) unsigned short Vs[64 * SQ];  // [d][key]
  __shared__ __align__(16) unsigned short Ps[64 * SQ];  // [qrow][key]
  __shared__ __align__(16) float Ss[64 * SS];
  __shared__ float rowm[64], rowl[64];

  const int tid  = threadIdx.x, lane = tid & 31, wave = tid >> 5;
  const int tr   = wave & 3, tc0 = (wave >> 2) * 2;
  const int qbase = blockIdx.x * 64;
  const int h = blockIdx.y, b = blockIdx.z;
  const size_t headoff = ((size_t)(b * H + h)) * NSEQ * DK;
  const unsigned short* Qh  = Qp + headoff;   // [n][d]
  const unsigned short* Kh  = Kp + headoff;   // [n][d]
  const unsigned short* Vth = Vt + headoff;   // [d][n]
  const float scale = 0.125f;                 // 1/sqrt(DK)

  // ---- load Q tile once ----
#if HAVE_TDM
  if (wave == 0)
    tdm_load_2d(Qh + (size_t)qbase * DK, (unsigned)(size_t)&Qs[0],
                /*tile*/ 64, 64, /*tensor*/ 64, NSEQ, /*stride*/ 64, 4, 3);
  __builtin_amdgcn_s_wait_tensorcnt((short)0);
#else
  {
    const int row = tid >> 2, off = (tid & 3) * 16;
    const unsigned short* src = Qh + (size_t)(qbase + row) * DK + off;
    *(u4v*)&Qs[row * SQ + off]     = *(const u4v*)(src);
    *(u4v*)&Qs[row * SQ + off + 8] = *(const u4v*)(src + 8);
  }
#endif
  if (tid < 64) { rowm[tid] = -3.0e38f; rowl[tid] = 0.f; }
  __syncthreads();

  // ---------------- pass 1: online max + sum ----------------
  for (int kt = 0; kt < NSEQ / 64; ++kt) {
#if HAVE_TDM
    if (wave == 0)
      tdm_load_2d(Kh + (size_t)(kt * 64) * DK, (unsigned)(size_t)&Ks[0],
                  64, 64, 64, NSEQ, 64, 4, 3);
    __builtin_amdgcn_s_wait_tensorcnt((short)0);
#else
    {
      const int row = tid >> 2, off = (tid & 3) * 16;
      const unsigned short* src = Kh + (size_t)(kt * 64 + row) * DK + off;
      *(u4v*)&Ks[row * SQ + off]     = *(const u4v*)(src);
      *(u4v*)&Ks[row * SQ + off + 8] = *(const u4v*)(src + 8);
    }
#endif
    __syncthreads();
    v8f s0 = {}, s1 = {};
#pragma unroll
    for (int ks = 0; ks < 64; ks += 32) {
      v16bf a  = load_a_frag(&Qs[(tr * 16) * SQ + ks], SQ, lane);
      v16bf b0 = load_b_frag(&Ks[(tc0 * 16) * SQ + ks], SQ, lane);
      v16bf b1 = load_b_frag(&Ks[((tc0 + 1) * 16) * SQ + ks], SQ, lane);
      s0 = wmma_bf16(a, b0, s0);
      s1 = wmma_bf16(a, b1, s1);
    }
    const int col = lane & 15, hi = (lane >> 4) & 1;
#pragma unroll
    for (int r = 0; r < 8; ++r) {
      Ss[(tr * 16 + r + 8 * hi) * SS + tc0 * 16 + col]       = s0[r] * scale;
      Ss[(tr * 16 + r + 8 * hi) * SS + (tc0 + 1) * 16 + col] = s1[r] * scale;
    }
    __syncthreads();
    if (tid < 64) {
      float m = rowm[tid], l = rowl[tid], tmax = -3.0e38f;
#pragma unroll 8
      for (int c = 0; c < 64; ++c) tmax = fmaxf(tmax, Ss[tid * SS + c]);
      const float nm = fmaxf(m, tmax);
      float sum = 0.f;
#pragma unroll 8
      for (int c = 0; c < 64; ++c) sum += __expf(Ss[tid * SS + c] - nm);
      rowl[tid] = l * __expf(m - nm) + sum;
      rowm[tid] = nm;
    }
    __syncthreads();
  }

  // ---------------- pass 2: emit P, accumulate A = P @ V ----------------
  v8f a0 = {}, a1 = {};
  for (int kt = 0; kt < NSEQ / 64; ++kt) {
#if HAVE_TDM
    if (wave == 0) {
      tdm_load_2d(Kh + (size_t)(kt * 64) * DK, (unsigned)(size_t)&Ks[0],
                  64, 64, 64, NSEQ, 64, 4, 3);
      tdm_load_2d(Vth + (size_t)(kt * 64), (unsigned)(size_t)&Vs[0],
                  64, 64, NSEQ, 64, NSEQ, 4, 3);   // rows are d, cols are n
    }
    __builtin_amdgcn_s_wait_tensorcnt((short)0);
#else
    {
      const int row = tid >> 2, off = (tid & 3) * 16;
      const unsigned short* srck = Kh + (size_t)(kt * 64 + row) * DK + off;
      *(u4v*)&Ks[row * SQ + off]     = *(const u4v*)(srck);
      *(u4v*)&Ks[row * SQ + off + 8] = *(const u4v*)(srck + 8);
      const unsigned short* srcv = Vth + (size_t)row * NSEQ + kt * 64 + off;
      *(u4v*)&Vs[row * SQ + off]     = *(const u4v*)(srcv);
      *(u4v*)&Vs[row * SQ + off + 8] = *(const u4v*)(srcv + 8);
    }
#endif
    __syncthreads();
    v8f s0 = {}, s1 = {};
#pragma unroll
    for (int ks = 0; ks < 64; ks += 32) {
      v16bf a  = load_a_frag(&Qs[(tr * 16) * SQ + ks], SQ, lane);
      v16bf b0 = load_b_frag(&Ks[(tc0 * 16) * SQ + ks], SQ, lane);
      v16bf b1 = load_b_frag(&Ks[((tc0 + 1) * 16) * SQ + ks], SQ, lane);
      s0 = wmma_bf16(a, b0, s0);
      s1 = wmma_bf16(a, b1, s1);
    }
    const int col = lane & 15, hi = (lane >> 4) & 1;
#pragma unroll
    for (int r = 0; r < 8; ++r) {
      Ss[(tr * 16 + r + 8 * hi) * SS + tc0 * 16 + col]       = s0[r] * scale;
      Ss[(tr * 16 + r + 8 * hi) * SS + (tc0 + 1) * 16 + col] = s1[r] * scale;
    }
    __syncthreads();
    { // normalized probabilities -> global attention (b128 stores) + LDS bf16
      const int row = tid >> 2, cg = (tid & 3) * 16;
      const float m = rowm[row], inv = 1.0f / rowl[row];
      float* attrow = att + ((size_t)(b * H + h) * NSEQ + qbase + row) * NSEQ + kt * 64 + cg;
      float p[16];
#pragma unroll
      for (int i = 0; i < 16; ++i) p[i] = __expf(Ss[row * SS + cg + i] - m) * inv;
#pragma unroll
      for (int i = 0; i < 4; ++i) *(f4v*)(attrow + i * 4) = *(f4v*)&p[i * 4];
      *(u4v*)&Ps[row * SQ + cg]     = pack_bf16_8(&p[0]);
      *(u4v*)&Ps[row * SQ + cg + 8] = pack_bf16_8(&p[8]);
    }
    __syncthreads();
#pragma unroll
    for (int ks = 0; ks < 64; ks += 32) {
      v16bf a  = load_a_frag(&Ps[(tr * 16) * SQ + ks], SQ, lane);
      v16bf b0 = load_b_frag(&Vs[(tc0 * 16) * SQ + ks], SQ, lane);
      v16bf b1 = load_b_frag(&Vs[((tc0 + 1) * 16) * SQ + ks], SQ, lane);
      a0 = wmma_bf16(a, b0, a0);
      a1 = wmma_bf16(a, b1, a1);
    }
    __syncthreads();
  }

  const int col = lane & 15, hi = (lane >> 4) & 1;
#pragma unroll
  for (int r = 0; r < 8; ++r) {
    const size_t qrow = (size_t)qbase + tr * 16 + r + 8 * hi;
    Abf[headoff + qrow * DK + tc0 * 16 + col]       = f32_to_bf16(a0[r]);
    Abf[headoff + qrow * DK + (tc0 + 1) * 16 + col] = f32_to_bf16(a1[r]);
  }
}

// ---------------------------------------------------------------------------
// Kernel 3a: Y = A(4096x1024 bf16)@Wa + ba + q. A-tile via TDM (pure copy).
// ---------------------------------------------------------------------------
__global__ __launch_bounds__(256) void outproj_kernel(
    const unsigned short* __restrict__ A, const float* __restrict__ Wa,
    const float* __restrict__ ba, const float* __restrict__ q,
    float* __restrict__ Y) {
  __shared__ __align__(16) unsigned short As[128 * SA];
  __shared__ __align__(16) unsigned short Bs[64 * SA];

  const int tid  = threadIdx.x;
  const int lane = tid & 31, wave = tid >> 5;
  const int tr0  = (wave & 3) * 2, tc0 = (wave >> 2) * 2;
  const int mbase = blockIdx.y * 128;
  const int nbase = blockIdx.x * 64;

  v8f acc00 = {}, acc01 = {}, acc10 = {}, acc11 = {};

  for (int k0 = 0; k0 < LDIM; k0 += 32) {
    __syncthreads();
#if HAVE_TDM
    if (wave == 0)
      tdm_load_2d(A + (size_t)mbase * LDIM + k0, (unsigned)(size_t)&As[0],
                  /*tile*/ 32, 128, /*tensor*/ LDIM, BS * NSEQ, /*stride*/ LDIM, 3, 3);
#else
    {
      const int row = tid >> 1, off = (tid & 1) * 16;
      const unsigned short* src = A + (size_t)(mbase + row) * LDIM + k0 + off;
      *(u4v*)&As[row * SA + off]     = *(const u4v*)(src);
      *(u4v*)&As[row * SA + off + 8] = *(const u4v*)(src + 8);
    }
#endif
    { // Wa tile 32x64 fp32, register transpose + convert
      const int col = tid & 63, kg = (tid >> 6) * 8;
      const float* src = Wa + (size_t)(k0 + kg) * LDIM + nbase + col;
      float t[8];
#pragma unroll
      for (int i = 0; i < 8; ++i) t[i] = src[(size_t)i * LDIM];
      *(u4v*)&Bs[col * SA + kg] = pack_bf16_8(t);
    }
    if (k0 + 32 < LDIM && tid < 64)
      __builtin_prefetch(Wa + (size_t)(k0 + 32 + (tid >> 1)) * LDIM + nbase + (tid & 1) * 32, 0, 1);
#if HAVE_TDM
    __builtin_amdgcn_s_wait_tensorcnt((short)0);
#endif
    __syncthreads();
    v16bf a0 = load_a_frag(&As[(tr0 * 16) * SA], SA, lane);
    v16bf a1 = load_a_frag(&As[((tr0 + 1) * 16) * SA], SA, lane);
    v16bf b0 = load_b_frag(&Bs[(tc0 * 16) * SA], SA, lane);
    v16bf b1 = load_b_frag(&Bs[((tc0 + 1) * 16) * SA], SA, lane);
    acc00 = wmma_bf16(a0, b0, acc00);
    acc01 = wmma_bf16(a0, b1, acc01);
    acc10 = wmma_bf16(a1, b0, acc10);
    acc11 = wmma_bf16(a1, b1, acc11);
  }

  const int col0 = lane & 15, hi = (lane >> 4) & 1;
  const v8f accs[4] = {acc00, acc01, acc10, acc11};
#pragma unroll
  for (int rt = 0; rt < 2; ++rt)
#pragma unroll
    for (int ct = 0; ct < 2; ++ct) {
      const v8f acc = accs[rt * 2 + ct];
      const int c = nbase + (tc0 + ct) * 16 + col0;
      const float bb = ba[c];
#pragma unroll
      for (int r = 0; r < 8; ++r) {
        const size_t grow = (size_t)mbase + (tr0 + rt) * 16 + r + 8 * hi;
        Y[grow * LDIM + c] = acc[r] + bb + q[grow * LDIM + c];
      }
    }
}

// ---------------------------------------------------------------------------
// Kernel 3b: row-wise LayerNorm
// ---------------------------------------------------------------------------
__global__ __launch_bounds__(256) void layernorm_kernel(
    const float* __restrict__ Y, const float* __restrict__ gamma,
    const float* __restrict__ beta, float* __restrict__ out) {
  const int row = blockIdx.x;
  const float* y = Y + (size_t)row * LDIM;
  const int c4 = threadIdx.x * 4;
  const f4v yv = *(const f4v*)(y + c4);
  float s  = yv[0] + yv[1] + yv[2] + yv[3];
  float s2 = yv[0] * yv[0] + yv[1] * yv[1] + yv[2] * yv[2] + yv[3] * yv[3];
#pragma unroll
  for (int off = 16; off > 0; off >>= 1) {
    s  += __shfl_down(s, off, 32);
    s2 += __shfl_down(s2, off, 32);
  }
  __shared__ float red[2][8];
  const int lane = threadIdx.x & 31, wave = threadIdx.x >> 5;
  if (lane == 0) { red[0][wave] = s; red[1][wave] = s2; }
  __syncthreads();
  if (wave == 0) {
    s  = (lane < 8) ? red[0][lane] : 0.f;
    s2 = (lane < 8) ? red[1][lane] : 0.f;
#pragma unroll
    for (int off = 4; off > 0; off >>= 1) {
      s  += __shfl_down(s, off, 32);
      s2 += __shfl_down(s2, off, 32);
    }
    if (lane == 0) { red[0][0] = s; red[1][0] = s2; }
  }
  __syncthreads();
  const float mu   = red[0][0] * (1.0f / LDIM);
  const float var  = red[1][0] * (1.0f / LDIM) - mu * mu;
  const float rinv = rsqrtf(var + EPS);
  const f4v gv = *(const f4v*)(gamma + c4);
  const f4v bv = *(const f4v*)(beta + c4);
  f4v ov;
#pragma unroll
  for (int i = 0; i < 4; ++i) ov[i] = gv[i] * (yv[i] - mu) * rinv + bv[i];
  *(f4v*)(out + (size_t)row * LDIM + c4) = ov;
}

// ---------------------------------------------------------------------------
extern "C" void kernel_launch(void* const* d_in, const int* in_sizes, int n_in,
                              void* d_out, int out_size, void* d_ws, size_t ws_size,
                              hipStream_t stream) {
  (void)in_sizes; (void)n_in; (void)out_size; (void)ws_size;
  const float* q     = (const float*)d_in[0];
  const float* k     = (const float*)d_in[1];
  const float* v     = (const float*)d_in[2];
  const float* Wq    = (const float*)d_in[3];
  const float* bq    = (const float*)d_in[4];
  const float* Wk    = (const float*)d_in[5];
  const float* bk    = (const float*)d_in[6];
  const float* Wv    = (const float*)d_in[7];
  const float* bv    = (const float*)d_in[8];
  const float* Wa    = (const float*)d_in[9];
  const float* ba    = (const float*)d_in[10];
  const float* gamma = (const float*)d_in[11];
  const float* beta  = (const float*)d_in[12];

  float* out = (float*)d_out;                          // (BS, N, L)
  float* att = out + (size_t)BS * NSEQ * LDIM;         // (BS, H, N, N)

  const size_t P = (size_t)BS * NSEQ * LDIM;           // 4 Mi elements
  unsigned short* wsu = (unsigned short*)d_ws;
  unsigned short* Qp  = wsu;                           // bf16 (b,h,n,d)
  unsigned short* Kp  = wsu + P;                       // bf16 (b,h,n,d)
  unsigned short* Vt  = wsu + 2 * P;                   // bf16 (b,h,d,n)
  unsigned short* Abf = wsu + 3 * P;                   // bf16 (b,h,n,d) flat
  float* Y = (float*)d_ws;                             // reuse Qp+Kp (dead by then)

  dim3 blk(256);
  proj_kernel<<<dim3(LDIM / 64, (BS * NSEQ) / 128, 3), blk, 0, stream>>>(
      q, k, v, Wq, bq, Wk, bk, Wv, bv, Qp, Kp, Vt);
  attn_kernel<<<dim3(NSEQ / 64, H, BS), blk, 0, stream>>>(Qp, Kp, Vt, att, Abf);
  outproj_kernel<<<dim3(LDIM / 64, (BS * NSEQ) / 128), blk, 0, stream>>>(Abf, Wa, ba, q, Y);
  layernorm_kernel<<<BS * NSEQ, blk, 0, stream>>>(Y, gamma, beta, out);
}
